// NLSPNModel_41094247088271
// MI455X (gfx1250) — matrix operand order repaired and checked by reading, not compiled, not run on hardware
//
#include <hip/hip_runtime.h>
#include <hip/hip_bf16.h>

#define BB 2
#define HH 240
#define WW 1216
#define CHG 64
#define NUM 8               // K_F*K_F - 1
#define HW (HH*WW)          // 291840
#define BHW (BB*HW)         // 583680
#define NOUT 24             // 3*NUM
#define KDIM 576            // CHG*9
#define NPAD 32
#define XG 76               // WW/16
#define NGROUPS (BB*HH*XG)  // 36480 (divisible by 8)
#define CONV_BLOCKS (NGROUPS/8)   // 4560
#define PIX_BLOCKS (BHW/256)      // 2280 (exact)
#define CPAD 72             // channel-padded stride for guidance tile (16B aligned)

// ---- LDS pool layout (bytes) ----
#define LDS_SW_OFF   0
#define LDS_SW_BYTES (KDIM*NPAD*2)              // 36864: packed bf16 weights
#define LDS_GW_OFF   (LDS_SW_OFF + LDS_SW_BYTES)
#define LDS_GW_BYTES (8*54*CPAD*2)              // 62208: 8 waves x guidance tile
#define LDS_WF_OFF   (LDS_GW_OFF + LDS_GW_BYTES)
#define LDS_WF_BYTES (NOUT*KDIM*4)              // 55296: raw f32 weights (TDM dest)
#define LDS_ACC_OFF  LDS_WF_OFF                 // overlap: used only after K loop
#define LDS_TOTAL    (LDS_WF_OFF + LDS_WF_BYTES) // 154368 B

typedef __attribute__((ext_vector_type(16))) __bf16 v16bf;
typedef __attribute__((ext_vector_type(8)))  __bf16 v8bf;
typedef __attribute__((ext_vector_type(8)))  float  v8f;
typedef __attribute__((ext_vector_type(4)))  unsigned int u32x4;
typedef __attribute__((ext_vector_type(8)))  int i32x8;
typedef __attribute__((ext_vector_type(4)))  int i32x4;

// Zero-padded bilinear sample from a single-batch (H,W) float plane.
__device__ __forceinline__ float bilin(const float* __restrict__ f, float ys, float xs)
{
    float y0f = floorf(ys), x0f = floorf(xs);
    float wy1 = ys - y0f, wx1 = xs - x0f;
    float wy0 = 1.f - wy1, wx0 = 1.f - wx1;
    int y0 = (int)y0f, x0 = (int)x0f;
    float r = 0.f;
    if (y0 >= 0 && y0 < HH) {
        const float* row = f + (size_t)y0 * WW;
        if (x0 >= 0     && x0 < WW)     r += wy0 * wx0 * row[x0];
        if (x0 + 1 >= 0 && x0 + 1 < WW) r += wy0 * wx1 * row[x0 + 1];
    }
    if (y0 + 1 >= 0 && y0 + 1 < HH) {
        const float* row = f + (size_t)(y0 + 1) * WW;
        if (x0 >= 0     && x0 < WW)     r += wy1 * wx0 * row[x0];
        if (x0 + 1 >= 0 && x0 + 1 < WW) r += wy1 * wx1 * row[x0 + 1];
    }
    return r;
}

// Combine two 8-lane bf16 LDS loads into one WMMA fragment register set.
__device__ __forceinline__ v16bf frag2(const __bf16* p0, const __bf16* p1)
{
    v8bf lo = *(const v8bf*)p0;
    v8bf hi = *(const v8bf*)p1;
    return __builtin_shufflevector(lo, hi, 0,1,2,3,4,5,6,7,8,9,10,11,12,13,14,15);
}

// Fused: implicit-GEMM 3x3 conv over guidance (bf16 WMMA, f32 accum) +
// offset/affinity post-processing. K order = tap*64 + channel so each
// 32-wide K step stays inside one tap -> contiguous ds_load_b128 fragments.
__global__ __launch_bounds__(256) void conv_oa_post(
    const float* __restrict__ g, const float* __restrict__ w,
    const float* __restrict__ bias, const float* __restrict__ conf,
    const float* __restrict__ aff_scale,
    float* __restrict__ out_off, float* __restrict__ out_aff)
{
    __shared__ __align__(16) char smem[LDS_TOTAL];
    __bf16* sW    = (__bf16*)(smem + LDS_SW_OFF);   // [k16][n(32)][s(16)] packed
    __bf16* gwAll = (__bf16*)(smem + LDS_GW_OFF);   // per-wave [pos(54)][c pad 72]
    float*  sWf32 = (float*)(smem + LDS_WF_OFF);    // raw w_oa via TDM
    float*  sAcc  = (float*)(smem + LDS_ACC_OFF);   // epilogue D tiles (overlaps)

    const int tid  = threadIdx.x;
    const int wave = tid >> 5;
    const int lane = tid & 31;

    // Per-wave pixel group: 16 consecutive x within one row.
    const int group = blockIdx.x * 8 + wave;
    const int b  = group / (HH * XG);
    const int r0 = group % (HH * XG);
    const int y  = r0 / XG;
    const int x0 = (r0 % XG) << 4;

    // Stage guidance halo tile, channel-innermost: gw[pos*CPAD + c], pos=row*18+col.
    __bf16* gw = gwAll + wave * 54 * CPAD;
    const float* gb = g + (size_t)b * CHG * HW;
    for (int i = lane; i < CHG * 54; i += 32) {
        int c   = i / 54;
        int pos = i % 54;
        int rr  = pos / 18;
        int cc  = pos % 18;
        int yy  = y + rr - 1;
        int xx  = x0 + cc - 1;
        float v = 0.f;
        if (yy >= 0 && yy < HH && xx >= 0 && xx < WW)
            v = gb[c * HW + yy * WW + xx];
        gw[pos * CPAD + c] = (__bf16)v;
    }

    // TDM: DMA the full f32 weight tensor (24*64*3*3 = 13824 elems) into LDS.
    if (wave == 0) {
        unsigned long long ga = (unsigned long long)(uintptr_t)w;
        unsigned int ldsa = (unsigned int)(uintptr_t)sWf32;
        u32x4 g0;
        g0[0] = 1u;                                   // count=1 (valid user D#)
        g0[1] = ldsa;                                 // lds_addr
        g0[2] = (unsigned int)ga;                     // global_addr[31:0]
        g0[3] = (unsigned int)((ga >> 32) & 0x01FFFFFFu) | 0x80000000u; // addr hi | type=2
        i32x8 g1;
        g1[0] = 0x00020000;                 // workgroup_mask=0, data_size=4B
        g1[1] = (int)(13824u << 16);        // tensor_dim0[15:0] at bits 63:48
        g1[2] = 0x00010000;                 // tensor_dim0 hi=0 | tensor_dim1=1
        g1[3] = (int)(13824u << 16);        // tensor_dim1 hi=0 | tile_dim0=13824
        g1[4] = 1;                          // tile_dim1=1, tile_dim2=0
        g1[5] = 13824;                      // tensor_dim0_stride
        g1[6] = 0;
        g1[7] = 0;
        i32x4 z4 = {0, 0, 0, 0};
        i32x8 z8 = {0, 0, 0, 0, 0, 0, 0, 0};
        __builtin_amdgcn_tensor_load_to_lds(g0, g1, z4, z4, z8, 0);
        __builtin_amdgcn_s_wait_tensorcnt(0);
    }
    __syncthreads();

    // Pack weights LDS f32 -> bf16 in exact B-fragment order:
    // sW[((k16*32)+n)*16 + s], k = k16*16 + s, t = k/64, c = k%64.
    for (int i = tid; i < KDIM * NPAD; i += 256) {
        int s   = i & 15;
        int n   = (i >> 4) & 31;
        int k16 = i >> 9;
        int k   = k16 * 16 + s;
        int t   = k >> 6;
        int c   = k & 63;
        float v = (n < NOUT) ? sWf32[n * KDIM + c * 9 + t] : 0.f;
        sW[i] = (__bf16)v;
    }
    __syncthreads();

    const int m  = lane & 15;         // M row (pixel in group) / N column
    const int hi = (lane >> 4) & 1;   // upper lane half
    const int aOff = m * CPAD + hi * 16;    // per-lane A base (elems)
    const int bOff = hi * 512 + m * 16;     // per-lane B base (elems)

    v8f acc0 = {};                    // N = 0..15
    v8f acc1 = {};                    // N = 16..31 (24..31 dead)
#pragma unroll
    for (int ks = 0; ks < 18; ++ks) { // K = 576 in steps of 32 (one tap / 2 steps)
        const int t     = ks >> 1;          // compile-time after unroll
        const int cbase = (ks & 1) << 5;
        const int apos  = ((t / 3) * 18 + (t % 3)) * CPAD + cbase; // const
        // A: slots0-15 -> c = cbase+hi*16+s (contiguous).
        v16bf a  = frag2(gw + apos + aOff, gw + apos + aOff + 8);
        // B: 16 contiguous k values per lane; b1 is N+16 (stride 16 elems).
        v16bf b0 = frag2(sW + ks * 1024 + bOff,       sW + ks * 1024 + bOff + 8);
        v16bf b1 = frag2(sW + ks * 1024 + bOff + 256, sW + ks * 1024 + bOff + 264);
        acc0 = __builtin_amdgcn_wmma_f32_16x16x32_bf16(
                   false, a, false, b0, (short)0, acc0, false, false);
        acc1 = __builtin_amdgcn_wmma_f32_16x16x32_bf16(
                   false, a, false, b1, (short)0, acc1, false, false);
    }
    __syncthreads();   // all LDS reads of sWf32/gw done before sAcc overlap reuse

    // Spill D (C/D layout: VGPR j -> M = j + 8*hi, N = lane&15) to LDS.
    float* sa = &sAcc[wave * 16 * NPAD];
#pragma unroll
    for (int j = 0; j < 8; ++j) {
        int mm = j + (hi ? 8 : 0);
        sa[mm * NPAD + m]      = acc0[j];
        sa[mm * NPAD + m + 16] = acc1[j];
    }
    __syncthreads();

    // Epilogue: lanes 0..15 each own one pixel (all 24 channels).
    if (lane < 16) {
        const int x = x0 + lane;
        float v[NOUT];
#pragma unroll
        for (int ch = 0; ch < NOUT; ++ch)
            v[ch] = sa[lane * NPAD + ch] + bias[ch];

        const float* cb = conf + (size_t)b * HW;
        const float ascale = aff_scale[0] + 1e-8f;
        float av[NUM];
        float asum = 0.f;
#pragma unroll
        for (int i = 0; i < NUM; ++i) {
            float oy = v[i], ox = v[NUM + i];
            float cs = bilin(cb, (float)y + oy, (float)x + ox);
            float af = tanhf(v[2 * NUM + i]) / ascale * cs;
            av[i] = af;
            asum += fabsf(af);
        }
        asum += 1e-4f;
        float sdiv = asum > 1.f ? asum : 1.f;
        float refv = 1.f;
#pragma unroll
        for (int i = 0; i < NUM; ++i) { av[i] /= sdiv; refv -= av[i]; }

        const size_t pbase = (size_t)y * WW + x;
#pragma unroll
        for (int kf = 0; kf < 9; ++kf) {
            float oy = 0.f, ox = 0.f, af = refv;
            if (kf != 4) {
                int i = kf < 4 ? kf : kf - 1;
                oy = v[i]; ox = v[NUM + i]; af = av[i];
            }
            out_off[((size_t)b * 18 + 2 * kf    ) * HW + pbase] = oy;
            out_off[((size_t)b * 18 + 2 * kf + 1) * HW + pbase] = ox;
            out_aff[((size_t)b * 9  + kf        ) * HW + pbase] = af;
        }
    }
}

// feat_blend = mask_fix ? feat_fix : feat_cur  (coalesced, once per step).
__global__ __launch_bounds__(256) void blend_fix(
    const float* __restrict__ cur, const float* __restrict__ fix,
    float* __restrict__ out)
{
    int i = blockIdx.x * 256 + threadIdx.x;
    float fx = fix[i];
    out[i] = fx > 0.f ? fx : cur[i];
}

// One propagation step: 9-tap bilinear gather weighted by affinities.
__global__ __launch_bounds__(256) void propagate(
    const float* __restrict__ fblend, const float* __restrict__ offs,
    const float* __restrict__ affs, float* __restrict__ out)
{
    int idx = blockIdx.x * 256 + threadIdx.x;
    int b = idx / HW;
    int p = idx % HW;
    int y = p / WW;
    int x = p % WW;
    const float* fb = fblend + (size_t)b * HW;
    const float* ob = offs + (size_t)b * 18 * HW + p;
    const float* ab = affs + (size_t)b * 9  * HW + p;

    __builtin_prefetch(fb + (size_t)y * WW + x, 0, 3);  // global_prefetch_b8

    float acc = 0.f;
#pragma unroll
    for (int kf = 0; kf < 9; ++kf) {
        float oy = ob[(size_t)(2 * kf)     * HW];
        float ox = ob[(size_t)(2 * kf + 1) * HW];
        float af = ab[(size_t)kf * HW];
        float ys = (float)(y + kf / 3 - 1) + oy;
        float xs = (float)(x + kf % 3 - 1) + ox;
        acc += af * bilin(fb, ys, xs);
    }
    out[idx] = acc;
}

extern "C" void kernel_launch(void* const* d_in, const int* in_sizes, int n_in,
                              void* d_out, int out_size, void* d_ws, size_t ws_size,
                              hipStream_t stream)
{
    (void)in_sizes; (void)n_in; (void)out_size; (void)ws_size;

    const float* feat_init = (const float*)d_in[0];
    const float* guidance  = (const float*)d_in[1];
    const float* confid    = (const float*)d_in[2];
    const float* feat_fix  = (const float*)d_in[3];
    const float* w_oa      = (const float*)d_in[4];
    const float* b_oa      = (const float*)d_in[5];
    const float* aff_scale = (const float*)d_in[6];

    float* out_feat = (float*)d_out;                  // (B,1,H,W)
    float* out_off  = out_feat + (size_t)BHW;         // (B,18,H,W)
    float* out_aff  = out_off + (size_t)BB * 18 * HW; // (B,9,H,W)

    float* blendbuf = (float*)d_ws;                   // 1 x BHW floats

    conv_oa_post<<<CONV_BLOCKS, 256, 0, stream>>>(
        guidance, w_oa, b_oa, confid, aff_scale, out_off, out_aff);

    const float* cur = feat_init;
    for (int it = 0; it < 18; ++it) {
        blend_fix<<<PIX_BLOCKS, 256, 0, stream>>>(cur, feat_fix, blendbuf);
        propagate<<<PIX_BLOCKS, 256, 0, stream>>>(blendbuf, out_off, out_aff,
                                                  out_feat);
        cur = out_feat;   // ping-pong through d_out feat slot
    }
}
